// NTM_16509854286497
// MI455X (gfx1250) — matrix operand order repaired and checked by reading, not compiled, not run on hardware
//
#include <hip/hip_runtime.h>
#include <cmath>

// ---------------------------------------------------------------------------
// NTM forward, MI455X (gfx1250, wave32, WMMA + TDM).
// B=1024, T=256, I=64, O=64, H=100, N=128, M=20.
//
// Persistent workgroup per 16-batch tile (64 blocks x 128 threads). All
// recurrent state in LDS (~241 KB < 320 KB/WGP). Matmuls via
// v_wmma_f32_16x16x32_bf16 on a shared A matrix [x|r|h], Kpad=192.
// x_t tiles are streamed into LDS by the Tensor Data Mover (double-buffered,
// DMA for x_{t+1} overlaps step-t compute; synced with s_wait_tensorcnt).
// ---------------------------------------------------------------------------

#define Bsz   1024
#define Tn    256
#define In    64
#define On    64
#define Hn    100
#define Nn    128
#define Mn    20
#define KPAD  192        // [ x(64) | r(20) | h(100) | 0(8) ]  -> 6 K-chunks of 32
#define KC    6
#define BT    16         // batch tile per block
#define EPSF  1e-8f

typedef __bf16 bf16_t;
typedef unsigned int u32;
typedef __attribute__((ext_vector_type(16))) __bf16 v16bf;
typedef __attribute__((ext_vector_type(8)))  __bf16 v8bf;
typedef __attribute__((ext_vector_type(8)))  float  v8f;
typedef __attribute__((ext_vector_type(4)))  u32    u32x4;
typedef __attribute__((ext_vector_type(8)))  u32    u32x8;

// packed weight rows (Npad): gates 400, write 80(66), read 32(26), out 64
#define NG_PAD 400
#define NW_PAD 80
#define NR_PAD 32
#define NO_PAD 64
#define WROWS  (NG_PAD + NW_PAD + NR_PAD + NO_PAD)   // 576

__device__ __forceinline__ float sigmf_(float x) { return 1.f / (1.f + __expf(-x)); }
__device__ __forceinline__ float softplusf_(float x) {
  return (x > 20.f) ? x : log1pf(__expf(x));
}

// ---------------------------------------------------------------------------
// Prep kernel: pack all weights (f32) into bf16 [row][KPAD] with the shared
// K layout: cols 0..83 = ci (x|r) inputs, cols 84..183 = h inputs, rest zero.
// For out_W: cols 84..183 = h part, cols 64..83 = read_vec part.
// ---------------------------------------------------------------------------
__global__ void ntm_pack_weights(const float* __restrict__ W_ih,
                                 const float* __restrict__ W_hh,
                                 const float* __restrict__ read_W,
                                 const float* __restrict__ write_W,
                                 const float* __restrict__ out_W,
                                 bf16_t* __restrict__ Wp) {
  int idx = blockIdx.x * blockDim.x + threadIdx.x;
  if (idx >= WROWS * KPAD) return;
  int r = idx / KPAD, k = idx % KPAD;
  float v = 0.f;
  if (r < NG_PAD) {                       // LSTM gates: ci @ W_ih^T + h @ W_hh^T
    if (k < 84)            v = W_ih[r * 84 + k];
    else if (k < 184)      v = W_hh[r * 100 + (k - 84)];
  } else if (r < NG_PAD + NW_PAD) {       // write head: h only
    int n = r - NG_PAD;
    if (n < 66 && k >= 84 && k < 184) v = write_W[n * 100 + (k - 84)];
  } else if (r < NG_PAD + NW_PAD + NR_PAD) {  // read head: h only
    int n = r - NG_PAD - NW_PAD;
    if (n < 26 && k >= 84 && k < 184) v = read_W[n * 100 + (k - 84)];
  } else {                                // output: [h, r]
    int n = r - NG_PAD - NW_PAD - NR_PAD;
    if (k >= 84 && k < 184)      v = out_W[n * 120 + (k - 84)];
    else if (k >= 64 && k < 84)  v = out_W[n * 120 + 100 + (k - 64)];
  }
  Wp[idx] = (bf16_t)v;
}

// ---------------------------------------------------------------------------
// WMMA fragment helpers (bf16 16x16x32, wave32).
// ---------------------------------------------------------------------------
__device__ __forceinline__ void load_afrags(const bf16_t* __restrict__ Ag,
                                            int lane, v16bf* af) {
  int m = lane & 15, kh = lane >> 4;
  const bf16_t* base = Ag + m * KPAD + kh * 16;
#pragma unroll
  for (int kc = 0; kc < KC; ++kc) {
    const bf16_t* ap = base + kc * 32;
    v8bf lo = *(const v8bf*)(ap);
    v8bf hi = *(const v8bf*)(ap + 8);
    v16bf a;
#pragma unroll
    for (int i = 0; i < 8; ++i) { a[i] = lo[i]; a[i + 8] = hi[i]; }
    af[kc] = a;
  }
}

__device__ __forceinline__ v8f wmma_tile(const v16bf* af,
                                         const bf16_t* __restrict__ W,
                                         int nt, int lane) {
  int nl = lane & 15, kh = lane >> 4;
  const bf16_t* wb = W + (size_t)(nt * 16 + nl) * KPAD + kh * 16;
  v8f acc = {};
#pragma unroll
  for (int kc = 0; kc < KC; ++kc) {
    v16bf b = *(const v16bf*)(wb + kc * 32);
    acc = __builtin_amdgcn_wmma_f32_16x16x32_bf16(
        /*neg_a=*/false, af[kc], /*neg_b=*/false, b,
        /*c_mod=*/(short)0, acc, /*reuse_a=*/false, /*reuse_b=*/false);
  }
  return acc;
}

// ---------------------------------------------------------------------------
// TDM: issue a 2D tensor_load_to_lds of a [16 x 64] f32 tile (row stride
// Tn*In elements) into LDS at byte offset lds_off. Wave-level op; issue from
// one wave only. D# built per CDNA5 ISA §8 (group0 4 SGPRs, group1 8 SGPRs).
// ---------------------------------------------------------------------------
__device__ __forceinline__ void tdm_issue_x(const float* gptr, u32 lds_off,
                                            u32x8 g1) {
  unsigned long long ga = (unsigned long long)(uintptr_t)gptr;
  u32x4 g0;
  g0[0] = 1u;                                        // count=1, user descriptor
  g0[1] = lds_off;                                   // lds_addr (bytes)
  g0[2] = (u32)ga;                                   // global_addr[31:0]
  g0[3] = (u32)((ga >> 32) & 0x01FFFFFFu) | (2u << 30); // addr[56:32] | type=2
  asm volatile("tensor_load_to_lds %0, %1" :: "s"(g0), "s"(g1) : "memory");
}

__device__ __forceinline__ u32x8 tdm_make_g1() {
  u32x8 g1;
  g1[0] = (2u << 16);          // workgroup_mask=0, data_size=2 (4 bytes)
  g1[1] = ((u32)In << 16);     // atomic_barrier_addr=0 | tensor_dim0.lo16 = 64
  g1[2] = ((u32)BT << 16);     // tensor_dim0.hi16=0 | tensor_dim1.lo16 = 16
  g1[3] = ((u32)In << 16);     // tensor_dim1.hi16=0 | tile_dim0 = 64
  g1[4] = (u32)BT;             // tile_dim1 = 16, tile_dim2 = 0
  g1[5] = (u32)(Tn * In);      // tensor_dim0_stride.lo32 = 16384 elements
  g1[6] = 0u;                  // stride0.hi16 = 0, stride1.lo16 = 0
  g1[7] = 0u;                  // stride1.hi32 = 0
  return g1;
}

// ---------------------------------------------------------------------------
// NTM addressing: content (cosine+softmax), interpolate, circular shift,
// sharpen, normalize. Thread n handles memory slot n for all 16 batch rows;
// 16 lanes do per-batch serial reductions. wprev updated in place.
// ---------------------------------------------------------------------------
__device__ void ntm_address(float* __restrict__ memb, float* __restrict__ wprev,
                            float* __restrict__ wtmp, float* __restrict__ kt,
                            float* __restrict__ scal,
                            const float* __restrict__ p, int pstride,
                            const float* __restrict__ hb, int tid) {
  if (tid < BT) {
    int b = tid;
    const float* pb = p + b * pstride;
    float beta  = softplusf_(pb[20] + hb[20]);
    float g     = sigmf_(pb[21] + hb[21]);
    float s0 = pb[22] + hb[22], s1 = pb[23] + hb[23], s2 = pb[24] + hb[24];
    float smx = fmaxf(s0, fmaxf(s1, s2));
    float e0 = __expf(s0 - smx), e1 = __expf(s1 - smx), e2 = __expf(s2 - smx);
    float si = 1.f / (e0 + e1 + e2);
    float gamma = 1.f + softplusf_(pb[25] + hb[25]);
    float nrm = 0.f;
#pragma unroll
    for (int m = 0; m < Mn; ++m) {
      float kv = tanhf(pb[m] + hb[m]);
      kt[b * Mn + m] = kv;
      nrm += kv * kv;
    }
    float* sc = scal + b * 12;
    sc[0] = beta; sc[1] = g; sc[2] = gamma;
    sc[3] = e0 * si; sc[4] = e1 * si; sc[5] = e2 * si;
    sc[6] = sqrtf(nrm);
  }
  __syncthreads();
  {
    int n = tid;
    for (int b = 0; b < BT; ++b) {
      const float* mrow = memb + (b * Nn + n) * Mn;
      const float* kb = kt + b * Mn;
      float dot = 0.f, mn = 0.f;
#pragma unroll
      for (int m = 0; m < Mn; ++m) { float mv = mrow[m]; dot += mv * kb[m]; mn += mv * mv; }
      const float* sc = scal + b * 12;
      wtmp[b * Nn + n] = sc[0] * dot / (sqrtf(mn) * sc[6] + EPSF);
    }
  }
  __syncthreads();
  if (tid < BT) {
    int b = tid;
    float mx = -1e30f;
    for (int n = 0; n < Nn; ++n) mx = fmaxf(mx, wtmp[b * Nn + n]);
    float s = 0.f;
    for (int n = 0; n < Nn; ++n) { float e = __expf(wtmp[b * Nn + n] - mx); wtmp[b * Nn + n] = e; s += e; }
    scal[b * 12 + 7] = 1.f / s;
  }
  __syncthreads();
  {
    int n = tid;
    for (int b = 0; b < BT; ++b) {
      const float* sc = scal + b * 12;
      float wc = wtmp[b * Nn + n] * sc[7];
      float g = sc[1];
      wtmp[b * Nn + n] = g * wc + (1.f - g) * wprev[b * Nn + n];
    }
  }
  __syncthreads();
  {
    int n = tid;
    int np = (n + 1) & (Nn - 1), nm = (n + Nn - 1) & (Nn - 1);
    for (int b = 0; b < BT; ++b) {
      const float* sc = scal + b * 12;
      float ws = sc[3] * wtmp[b * Nn + np] + sc[4] * wtmp[b * Nn + n] + sc[5] * wtmp[b * Nn + nm];
      wprev[b * Nn + n] = __powf(fmaxf(ws, 1e-30f), sc[2]);
    }
  }
  __syncthreads();
  if (tid < BT) {
    int b = tid;
    float s = 0.f;
    for (int n = 0; n < Nn; ++n) s += wprev[b * Nn + n];
    scal[b * 12 + 7] = 1.f / (s + EPSF);
  }
  __syncthreads();
  {
    int n = tid;
    for (int b = 0; b < BT; ++b) wprev[b * Nn + n] *= scal[b * 12 + 7];
  }
  __syncthreads();
}

// ---------------------------------------------------------------------------
// Dynamic LDS layout (xstage double-buffer first so its byte offset is 0).
// ---------------------------------------------------------------------------
#define SM_FLOATS (2*BT*In + BT*Nn*Mn + BT*400 + BT*Hn + 3*BT*Nn + BT*NW_PAD + BT*NR_PAD + 3*BT*Mn + BT*12)
#define SMEM_BYTES (SM_FLOATS*4 + BT*KPAD*2)

__global__ void __launch_bounds__(128)
ntm_forward(const float* __restrict__ x_seq,
            const float* __restrict__ b_ih, const float* __restrict__ b_hh,
            const float* __restrict__ read_b, const float* __restrict__ write_b,
            const float* __restrict__ out_b,
            const bf16_t* __restrict__ Wp,
            float* __restrict__ out) {
  extern __shared__ __align__(16) char smem_raw[];
  float* xstage = (float*)smem_raw;         // [2][16][64] TDM staging (off 0)
  float* memb  = xstage + 2 * BT * In;      // [16][128][20] NTM memory
  float* gates = memb  + BT * Nn * Mn;      // [16][400]
  float* cst   = gates + BT * 400;          // [16][100] LSTM cell
  float* w_w   = cst   + BT * Hn;           // [16][128] write weights
  float* w_r   = w_w   + BT * Nn;           // [16][128] read weights
  float* wtmp  = w_r   + BT * Nn;           // [16][128] scratch
  float* pw    = wtmp  + BT * Nn;           // [16][80] write-head params
  float* pr    = pw    + BT * NW_PAD;       // [16][32] read-head params
  float* kt    = pr    + BT * NR_PAD;       // [16][20] addressing key
  float* ew    = kt    + BT * Mn;           // [16][20] erase
  float* aw    = ew    + BT * Mn;           // [16][20] add
  float* scal  = aw    + BT * Mn;           // [16][12] per-batch scalars
  bf16_t* Ag   = (bf16_t*)(scal + BT * 12); // [16][192] shared A matrix [x|r|h]

  const int tid  = threadIdx.x;
  const int lane = tid & 31;
  const int wave = __builtin_amdgcn_readfirstlane(tid >> 5);  // scalar wave id
  const int batch0 = blockIdx.x * BT;

  const bf16_t* Wg = Wp;
  const bf16_t* Ww = Wp + (size_t)NG_PAD * KPAD;
  const bf16_t* Wr = Ww + (size_t)NW_PAD * KPAD;
  const bf16_t* Wo = Wr + (size_t)NR_PAD * KPAD;

  // ---- initial state ----
  for (int i = tid; i < BT * KPAD; i += 128) Ag[i] = (bf16_t)0.f;      // x=r=h=0
  for (int i = tid; i < BT * Hn; i += 128) cst[i] = 0.f;
  for (int i = tid; i < BT * Nn; i += 128) { w_r[i] = 1.f / Nn; w_w[i] = 1.f / Nn; }
  for (int i = tid; i < BT * Nn * Mn; i += 128) memb[i] = 1e-6f;

  // ---- TDM prologue: kick off DMA of x_0 into staging buffer 0 ----
  const u32x8 g1 = tdm_make_g1();
  const float* xbase = x_seq + (size_t)batch0 * Tn * In;
  if (wave == 0) tdm_issue_x(xbase /* t=0 */, 0u, g1);
  __syncthreads();

  v16bf af[KC];

  for (int t = 0; t < Tn; ++t) {
    // ---- software-pipelined TDM: issue x_{t+1}, wait for x_t ----
    if (wave == 0) {
      if (t + 1 < Tn) {
        tdm_issue_x(xbase + (size_t)(t + 1) * In,
                    (u32)(((t + 1) & 1) * BT * In * 4), g1);
        __builtin_amdgcn_s_wait_tensorcnt(1);   // x_t landed (in-order)
      } else {
        __builtin_amdgcn_s_wait_tensorcnt(0);
      }
    }
    __syncthreads();

    // ---- convert staged x_t (f32) into A cols [0,64) (bf16) ----
    {
      const float* xs = xstage + (t & 1) * BT * In;
      for (int i = tid; i < BT * In; i += 128) {
        int b = i >> 6, k = i & 63;
        Ag[b * KPAD + k] = (bf16_t)xs[i];
      }
    }
    __syncthreads();

    // ---- LSTM gates: [16x400] = A[16x192] @ Wg^T ----
    load_afrags(Ag, lane, af);
    for (int nt = wave; nt < NG_PAD / 16; nt += 4) {
      v8f acc = wmma_tile(af, Wg, nt, lane);
      int n = nt * 16 + (lane & 15);
#pragma unroll
      for (int r = 0; r < 8; ++r) {
        int mrow = (lane >> 4) * 8 + r;
        gates[mrow * 400 + n] = acc[r];
      }
    }
    __syncthreads();

    // ---- LSTM elementwise: update c, h (h -> A cols [84,184)) ----
    for (int i = tid; i < BT * Hn; i += 128) {
      int b = i / Hn, j = i % Hn;
      const float* gb = gates + b * 400;
      float ig = gb[j]       + b_ih[j]       + b_hh[j];
      float fg = gb[100 + j] + b_ih[100 + j] + b_hh[100 + j];
      float gg = gb[200 + j] + b_ih[200 + j] + b_hh[200 + j];
      float og = gb[300 + j] + b_ih[300 + j] + b_hh[300 + j];
      float cv = sigmf_(fg) * cst[b * Hn + j] + sigmf_(ig) * tanhf(gg);
      cst[b * Hn + j] = cv;
      Ag[b * KPAD + 84 + j] = (bf16_t)(sigmf_(og) * tanhf(cv));
    }
    __syncthreads();

    // ---- head projections: p_w[16x80], p_r[16x32] from new h ----
    load_afrags(Ag, lane, af);
    for (int tt = wave; tt < 7; tt += 4) {
      const bf16_t* W = (tt < 5) ? Ww : Wr;
      int nt = (tt < 5) ? tt : tt - 5;
      v8f acc = wmma_tile(af, W, nt, lane);
      float* dst = (tt < 5) ? pw : pr;
      int stride = (tt < 5) ? NW_PAD : NR_PAD;
      int n = nt * 16 + (lane & 15);
#pragma unroll
      for (int r = 0; r < 8; ++r) {
        int mrow = (lane >> 4) * 8 + r;
        dst[mrow * stride + n] = acc[r];
      }
    }
    __syncthreads();

    // ---- write head addressing ----
    ntm_address(memb, w_w, wtmp, kt, scal, pw, NW_PAD, write_b, tid);

    // ---- erase / add vectors ----
    for (int i = tid; i < BT * Mn; i += 128) {
      int b = i / Mn, m = i % Mn;
      ew[i] = sigmf_(pw[b * NW_PAD + 26 + m] + write_b[26 + m]);
      aw[i] = tanhf(pw[b * NW_PAD + 46 + m] + write_b[46 + m]);
    }
    __syncthreads();

    // ---- memory update ----
    {
      int n = tid;
      for (int b = 0; b < BT; ++b) {
        float wv = w_w[b * Nn + n];
        float* mrow = memb + (b * Nn + n) * Mn;
        const float* eb = ew + b * Mn;
        const float* ab = aw + b * Mn;
#pragma unroll
        for (int m = 0; m < Mn; ++m)
          mrow[m] = mrow[m] * (1.f - wv * eb[m]) + wv * ab[m];
      }
    }
    __syncthreads();

    // ---- read head addressing (on updated memory) ----
    ntm_address(memb, w_r, wtmp, kt, scal, pr, NR_PAD, read_b, tid);

    // ---- read vector -> A cols [64,84) ----
    for (int i = tid; i < BT * Mn; i += 128) {
      int b = i / Mn, m = i % Mn;
      float s = 0.f;
      for (int n = 0; n < Nn; ++n) s += w_r[b * Nn + n] * memb[(b * Nn + n) * Mn + m];
      Ag[b * KPAD + 64 + m] = (bf16_t)s;
    }
    __syncthreads();

    // ---- output: [16x64] = [h,r] @ Wo^T + out_b, stored to global ----
    load_afrags(Ag, lane, af);
    for (int nt = wave; nt < NO_PAD / 16; nt += 4) {
      v8f acc = wmma_tile(af, Wo, nt, lane);
      int n = nt * 16 + (lane & 15);
#pragma unroll
      for (int r = 0; r < 8; ++r) {
        int mrow = (lane >> 4) * 8 + r;
        out[((size_t)(batch0 + mrow) * Tn + t) * On + n] = acc[r] + out_b[n];
      }
    }
    __syncthreads();  // protect Ag before next-step x overwrite
  }
}

// ---------------------------------------------------------------------------
extern "C" void kernel_launch(void* const* d_in, const int* in_sizes, int n_in,
                              void* d_out, int out_size, void* d_ws, size_t ws_size,
                              hipStream_t stream) {
  const float* x_seq   = (const float*)d_in[0];
  const float* W_ih    = (const float*)d_in[1];
  const float* W_hh    = (const float*)d_in[2];
  const float* b_ih    = (const float*)d_in[3];
  const float* b_hh    = (const float*)d_in[4];
  const float* read_W  = (const float*)d_in[5];
  const float* read_b  = (const float*)d_in[6];
  const float* write_W = (const float*)d_in[7];
  const float* write_b = (const float*)d_in[8];
  const float* out_W   = (const float*)d_in[9];
  const float* out_b   = (const float*)d_in[10];
  float* out = (float*)d_out;
  bf16_t* Wp = (bf16_t*)d_ws;   // 576*192 bf16 = 216 KB packed weights

  (void)hipFuncSetAttribute((const void*)ntm_forward,
                            hipFuncAttributeMaxDynamicSharedMemorySize, SMEM_BYTES);

  ntm_pack_weights<<<(WROWS * KPAD + 255) / 256, 256, 0, stream>>>(
      W_ih, W_hh, read_W, write_W, out_W, Wp);

  ntm_forward<<<Bsz / BT, 128, SMEM_BYTES, stream>>>(
      x_seq, b_ih, b_hh, read_b, write_b, out_b, Wp, out);
}
